// MultiHeadAttention2D_72602127171604
// MI455X (gfx1250) — compile-verified, matrix operand order checked by reading, not run on hardware
//
#include <hip/hip_runtime.h>

#define BB 8
#define TT 512
#define DD 256
#define HH 8

typedef __attribute__((ext_vector_type(16))) _Float16 v16h;
typedef __attribute__((ext_vector_type(8)))  _Float16 v8h;
typedef __attribute__((ext_vector_type(8)))  float    v8f;

// ---------------------------------------------------------------------------
// CDNA5 async global->LDS copy (bypasses VGPRs, tracked by ASYNCcnt).
// lds_off = workgroup-relative LDS byte address (low 32 bits of the generic
// pointer to a __shared__ object), gaddr = 64-bit global address.
// ---------------------------------------------------------------------------
__device__ __forceinline__ void async_b128(unsigned int lds_off,
                                           const void* gaddr) {
  asm volatile("global_load_async_to_lds_b128 %0, %1, off"
               :
               : "v"(lds_off), "v"((unsigned long long)(uintptr_t)gaddr)
               : "memory");
}

template <int N>
__device__ __forceinline__ void wait_asynccnt() {
  asm volatile("s_wait_asynccnt %0" : : "n"(N) : "memory");
}

// ---------------------------------------------------------------------------
// Conv for q/k/v: 5x5, Cin=1, Cout=8, SAME padding. Writes f16 features in
// both [B,H,T,D] and transposed [B,H,D,T] layouts so every attention einsum
// becomes an NT GEMM (C = A * B^T with A,B row-major over K).
// ---------------------------------------------------------------------------
__global__ __launch_bounds__(256)
void conv_qkv(const float* __restrict__ q, const float* __restrict__ v,
              const float* __restrict__ wq, const float* __restrict__ bq,
              const float* __restrict__ wv, const float* __restrict__ bv,
              const float* __restrict__ wk, const float* __restrict__ bk,
              _Float16* __restrict__ fq, _Float16* __restrict__ fqT,
              _Float16* __restrict__ fk, _Float16* __restrict__ fkT,
              _Float16* __restrict__ fv, _Float16* __restrict__ fvT) {
  int idx = blockIdx.x * 256 + threadIdx.x;
  int d = idx & (DD - 1);
  int t = (idx >> 8) & (TT - 1);
  int b = idx >> 17;

  float aq[HH], ak[HH], av[HH];
#pragma unroll
  for (int h = 0; h < HH; ++h) { aq[h] = bq[h]; ak[h] = bk[h]; av[h] = bv[h]; }

#pragma unroll
  for (int i = 0; i < 5; ++i) {
    int tt = t + i - 2;
    if (tt < 0 || tt >= TT) continue;
#pragma unroll
    for (int j = 0; j < 5; ++j) {
      int dd = d + j - 2;
      if (dd < 0 || dd >= DD) continue;
      size_t off = ((size_t)b * TT + tt) * DD + dd;
      float qv = q[off];
      float vv = v[off];
      int wb = (i * 5 + j) * HH;
#pragma unroll
      for (int h = 0; h < HH; ++h) {
        aq[h] += qv * wq[wb + h];
        ak[h] += vv * wk[wb + h];
        av[h] += vv * wv[wb + h];
      }
    }
  }
#pragma unroll
  for (int h = 0; h < HH; ++h) {
    size_t n  = (((size_t)b * HH + h) * TT + t) * DD + d;
    size_t tr = (((size_t)b * HH + h) * DD + d) * TT + t;
    fq[n] = (_Float16)aq[h]; fqT[tr] = (_Float16)aq[h];
    fk[n] = (_Float16)ak[h]; fkT[tr] = (_Float16)ak[h];
    fv[n] = (_Float16)av[h]; fvT[tr] = (_Float16)av[h];
  }
}

// ---------------------------------------------------------------------------
// Batched NT GEMM: C[bz] = A[bz] * B[bz]^T
//   A: [M,K] row-major (lda=K), B: [N,K] row-major (ldb=K), C: [M,N] (ldc=N)
// Block tile 128x128, 8 waves, each wave 32x64 (2x4 WMMA f32_16x16x32_f16).
// K staged through double-buffered LDS (chunk=32) filled by CDNA5 async
// global->LDS copies; next tile's copies overlap current tile's WMMA work.
// ---------------------------------------------------------------------------
template <bool STORE_F16>
__global__ __launch_bounds__(256)
void gemm_nt(const _Float16* __restrict__ A, const _Float16* __restrict__ B,
             void* __restrict__ Cv, int M, int N, int K) {
  __shared__ _Float16 lA[2][128][40];   // pad rows to 40 halves (16B aligned)
  __shared__ _Float16 lB[2][128][40];

  const int tid  = threadIdx.x;
  const int lane = tid & 31;
  const int wave = tid >> 5;
  const int bz   = blockIdx.z;

  const _Float16* Ab = A + (size_t)bz * M * K;
  const _Float16* Bb = B + (size_t)bz * N * K;

  const int rowBase = blockIdx.y * 128;
  const int colBase = blockIdx.x * 128;
  const int wRow = (wave >> 1) * 32;   // 4 wave-rows
  const int wCol = (wave & 1) * 64;    // 2 wave-cols
  const int r  = lane & 15;
  const int hi = lane >> 4;            // half-wave selects K sub-block

  v8f acc[2][4];
#pragma unroll
  for (int i = 0; i < 2; ++i)
#pragma unroll
    for (int j = 0; j < 4; ++j)
#pragma unroll
      for (int e = 0; e < 8; ++e) acc[i][j][e] = 0.0f;

  // Per-thread copy slice: row = tid>>1, halves [seg, seg+16)
  const int ldRow = tid >> 1;
  const int ldSeg = (tid & 1) * 16;
  const _Float16* aRow = Ab + (size_t)(rowBase + ldRow) * K + ldSeg;
  const _Float16* bRow = Bb + (size_t)(colBase + ldRow) * K + ldSeg;
  unsigned int lAoff[2], lBoff[2];
#pragma unroll
  for (int buf = 0; buf < 2; ++buf) {
    lAoff[buf] = (unsigned int)(uintptr_t)&lA[buf][ldRow][ldSeg];
    lBoff[buf] = (unsigned int)(uintptr_t)&lB[buf][ldRow][ldSeg];
  }

  auto issue_tile = [&](int buf, int k0) {
    const _Float16* a = aRow + k0;
    const _Float16* b = bRow + k0;
    async_b128(lAoff[buf],      a);       // halves [seg, seg+8)
    async_b128(lAoff[buf] + 16, a + 8);   // halves [seg+8, seg+16)
    async_b128(lBoff[buf],      b);
    async_b128(lBoff[buf] + 16, b + 8);
  };

  issue_tile(0, 0);   // prologue

  const int nIter = K >> 5;
  for (int it = 0; it < nIter; ++it) {
    const int cur = it & 1;
    if (it + 1 < nIter) {               // uniform branch: EXEC stays all-ones
      issue_tile(cur ^ 1, (it + 1) << 5);
      wait_asynccnt<4>();               // current tile done; next 4 in flight
    } else {
      wait_asynccnt<0>();
    }
    __syncthreads();                    // all waves' slices visible in LDS

    // Fragment layout (16-bit A operand, wave32): lane<16 holds row r with
    // K[0..7],K[16..23]; lane>=16 holds row r with K[8..15],K[24..31].
    v16h af[2], bf[4];
#pragma unroll
    for (int i = 0; i < 2; ++i) {
      int row = wRow + i * 16 + r;
      v8h lo  = *(const v8h*)&lA[cur][row][hi * 8];
      v8h hiv = *(const v8h*)&lA[cur][row][16 + hi * 8];
      af[i] = __builtin_shufflevector(lo, hiv, 0, 1, 2, 3, 4, 5, 6, 7,
                                      8, 9, 10, 11, 12, 13, 14, 15);
    }
#pragma unroll
    for (int j = 0; j < 4; ++j) {
      int row = wCol + j * 16 + r;
      v8h lo  = *(const v8h*)&lB[cur][row][hi * 8];
      v8h hiv = *(const v8h*)&lB[cur][row][16 + hi * 8];
      bf[j] = __builtin_shufflevector(lo, hiv, 0, 1, 2, 3, 4, 5, 6, 7,
                                      8, 9, 10, 11, 12, 13, 14, 15);
    }
#pragma unroll
    for (int i = 0; i < 2; ++i)
#pragma unroll
      for (int j = 0; j < 4; ++j)
        acc[i][j] = __builtin_amdgcn_wmma_f32_16x16x32_f16(
            false, af[i], false, bf[j], (short)0, acc[i][j], false, false);

    __syncthreads();   // reads done before this buffer is refilled next round
  }

  // C layout: VGPR p -> row p + (lane<16 ? 0 : 8), col = lane&15
#pragma unroll
  for (int i = 0; i < 2; ++i)
#pragma unroll
    for (int j = 0; j < 4; ++j) {
      int row0 = rowBase + wRow + i * 16 + hi * 8;
      int col  = colBase + wCol + j * 16 + r;
#pragma unroll
      for (int p = 0; p < 8; ++p) {
        size_t o = (size_t)(row0 + p) * N + col;
        if (STORE_F16)
          ((_Float16*)Cv + (size_t)bz * M * N)[o] = (_Float16)acc[i][j][p];
        else
          ((float*)Cv + (size_t)bz * M * N)[o] = acc[i][j][p];
      }
    }
}

// ---------------------------------------------------------------------------
// In-place row softmax on f16 scores (f32 math). rowLen in {256, 512}.
// ---------------------------------------------------------------------------
__global__ __launch_bounds__(256)
void softmax_inplace_f16(_Float16* __restrict__ S, int rowLen) {
  __shared__ float red[256];
  const int tid = threadIdx.x;
  _Float16* p = S + (size_t)blockIdx.x * rowLen;
  const int n = rowLen >> 8;   // elems per thread: 1 or 2

  float vv[2];
  float m = -3.0e38f;
  for (int i = 0; i < n; ++i) {
    vv[i] = (float)p[tid + 256 * i];
    m = fmaxf(m, vv[i]);
  }
  red[tid] = m; __syncthreads();
  for (int s = 128; s > 0; s >>= 1) {
    if (tid < s) red[tid] = fmaxf(red[tid], red[tid + s]);
    __syncthreads();
  }
  m = red[0]; __syncthreads();

  float sum = 0.0f;
  for (int i = 0; i < n; ++i) { vv[i] = __expf(vv[i] - m); sum += vv[i]; }
  red[tid] = sum; __syncthreads();
  for (int s = 128; s > 0; s >>= 1) {
    if (tid < s) red[tid] += red[tid + s];
    __syncthreads();
  }
  const float inv = 1.0f / red[0];
  for (int i = 0; i < n; ++i) p[tid + 256 * i] = (_Float16)(vv[i] * inv);
}

// ---------------------------------------------------------------------------
// Output conv: 5x5, Cin=16 (out_t heads 0..7, out_f heads 8..15), Cout=1.
// Ot/Of are [B,H,T,D] f32.
// ---------------------------------------------------------------------------
__global__ __launch_bounds__(256)
void out_conv(const float* __restrict__ Ot, const float* __restrict__ Of,
              const float* __restrict__ wo, const float* __restrict__ bo,
              float* __restrict__ out) {
  int idx = blockIdx.x * 256 + threadIdx.x;
  int d = idx & (DD - 1);
  int t = (idx >> 8) & (TT - 1);
  int b = idx >> 17;

  float acc = bo[0];
#pragma unroll
  for (int i = 0; i < 5; ++i) {
    int tt = t + i - 2;
    if (tt < 0 || tt >= TT) continue;
#pragma unroll
    for (int j = 0; j < 5; ++j) {
      int dd = d + j - 2;
      if (dd < 0 || dd >= DD) continue;
      const float* wp = wo + (i * 5 + j) * 16;   // wo[i][j][c][0]
#pragma unroll
      for (int h = 0; h < HH; ++h) {
        size_t o = (((size_t)b * HH + h) * TT + tt) * DD + dd;
        acc += Ot[o] * wp[h];
        acc += Of[o] * wp[8 + h];
      }
    }
  }
  out[((size_t)b * TT + t) * DD + d] = acc;
}

// ---------------------------------------------------------------------------
extern "C" void kernel_launch(void* const* d_in, const int* in_sizes, int n_in,
                              void* d_out, int out_size, void* d_ws, size_t ws_size,
                              hipStream_t stream) {
  (void)in_sizes; (void)n_in; (void)out_size; (void)ws_size;

  const float* q  = (const float*)d_in[0];
  const float* v  = (const float*)d_in[1];
  const float* wq = (const float*)d_in[2];
  const float* bq = (const float*)d_in[3];
  const float* wv = (const float*)d_in[4];
  const float* bv = (const float*)d_in[5];
  const float* wk = (const float*)d_in[6];
  const float* bk = (const float*)d_in[7];
  const float* wo = (const float*)d_in[8];
  const float* bo = (const float*)d_in[9];
  float* out = (float*)d_out;

  // Workspace layout (200 MB total):
  const size_t F = (size_t)BB * HH * TT * DD;        // 8,388,608 feature elems
  _Float16* fq  = (_Float16*)d_ws;                    // 6 x F halves
  _Float16* fk  = fq  + F;
  _Float16* fv  = fk  + F;
  _Float16* fqT = fv  + F;
  _Float16* fkT = fqT + F;
  _Float16* fvT = fkT + F;
  _Float16* St  = fvT + F;                            // B*H*T*T halves
  _Float16* Sf  = St + (size_t)BB * HH * TT * TT;     // B*H*D*D halves
  float*    Ot  = (float*)(Sf + (size_t)BB * HH * DD * DD);  // F f32
  float*    Of  = Ot + F;                                    // F f32

  dim3 blk(256);
  const int BH = BB * HH;

  // 1) q/k/v convs -> f16 features (normal + transposed layouts)
  conv_qkv<<<dim3((BB * TT * DD) / 256), blk, 0, stream>>>(
      q, v, wq, bq, wv, bv, wk, bk, fq, fqT, fk, fkT, fv, fvT);

  // 2) Temporal scores: St = fq * fk^T   (M=N=T=512, K=D=256), per (b,h)
  gemm_nt<true><<<dim3(4, 4, BH), blk, 0, stream>>>(fq, fk, St, TT, TT, DD);

  // 3) Row softmax over s (rowLen=T)
  softmax_inplace_f16<<<dim3(BH * TT), blk, 0, stream>>>(St, TT);

  // 4) Ot = P * fv = P * (fvT)^T   (M=T, N=D, K=T)
  gemm_nt<false><<<dim3(2, 4, BH), blk, 0, stream>>>(St, fvT, Ot, TT, DD, TT);

  // 5) Frequency scores: Sf = fqT * (fkT)^T   (M=N=D=256, K=T=512)
  gemm_nt<true><<<dim3(2, 2, BH), blk, 0, stream>>>(fqT, fkT, Sf, DD, DD, TT);

  // 6) Row softmax over e (rowLen=D)
  softmax_inplace_f16<<<dim3(BH * DD), blk, 0, stream>>>(Sf, DD);

  // 7) Of = fv * Af^T   (M=T, N=D, K=D)
  gemm_nt<false><<<dim3(2, 4, BH), blk, 0, stream>>>(fv, Sf, Of, TT, DD, DD);

  // 8) Output conv over concat([Ot, Of]) channels
  out_conv<<<dim3((BB * TT * DD) / 256), blk, 0, stream>>>(Ot, Of, wo, bo, out);
}
